// MLAAttention_12902081757317
// MI455X (gfx1250) — compile-verified, matrix operand order checked by reading, not run on hardware
//
#include <hip/hip_runtime.h>
#include <hip/hip_bf16.h>

// ---------------------------------------------------------------------------
// MLA attention for gfx1250 (MI455X). All GEMMs + attention matmuls via
// v_wmma_f32_16x16x32_bf16 (wave32 WMMA), fp32 accumulate, bf16 intermediates.
// K-tile staging in the flash kernel uses GLOBAL_LOAD_ASYNC_TO_LDS_B128
// (ASYNCcnt-tracked DMA into LDS) with the toolchain's v4i* signature.
// ---------------------------------------------------------------------------

typedef __bf16 bf16_t;
typedef __attribute__((ext_vector_type(16))) __bf16 v16bf;
typedef __attribute__((ext_vector_type(8)))  float  v8f;
typedef int v4i_t __attribute__((vector_size(4 * sizeof(int))));

#define AS_GLOBAL __attribute__((address_space(1)))
#define AS_LOCAL  __attribute__((address_space(3)))

#if defined(__has_builtin)
#if __has_builtin(__builtin_amdgcn_global_load_async_to_lds_b128) && \
    __has_builtin(__builtin_amdgcn_s_wait_asynccnt)
#define USE_ASYNC_LDS 1
#endif
#endif

constexpr int kHID = 2048;
constexpr int kH   = 16;     // heads
constexpr int kKV  = 4;      // kv heads
constexpr int kHD  = 128;    // latent head dim
constexpr int kOD  = 128;    // original head dim
constexpr int kLAT = 512;    // latent dim
constexpr int kB   = 2;
constexpr int kT   = 2048;

// ---------------------------------------------------------------------------
// Fragment load, CDNA5 16-bit A/B layout (ISA 7.12.2):
//   lane l holds row (l&15); VGPRs 0..3 hold K = half*8 + {0..7},
//   VGPRs 4..7 hold K = half*8 + 16 + {0..7}, half = l>>4.
// Two contiguous 16-byte chunks per lane -> vectorizes to ds_load_b128.
// ---------------------------------------------------------------------------
__device__ inline v16bf load_frag(const bf16_t* base, int ld, int row, int col,
                                  int lane) {
  const bf16_t* p = base + (size_t)(row + (lane & 15)) * ld + col +
                    ((lane >> 4) << 3);
  v16bf f;
#pragma unroll
  for (int i = 0; i < 8; ++i) f[i] = p[i];
#pragma unroll
  for (int i = 0; i < 8; ++i) f[8 + i] = p[16 + i];
  return f;
}

__device__ inline v8f wmma_bf16(v16bf a, v16bf b, v8f c) {
  // (neg_a, A, neg_b, B, c_mod, C, reuse_a, reuse_b)
  return __builtin_amdgcn_wmma_f32_16x16x32_bf16(false, a, false, b, (short)0,
                                                 c, false, false);
}

// ---------------------------------------------------------------------------
// Tiled GEMM: C[M,N] = A[M,K] * op(B) + bias, op(B) = B^T if B_NK (B is [N,K])
// else B (B is [K,N]). 128x128 C-tile per block (8 waves, each 32x64 = 8 WMMA
// accumulators). A source fp32 or bf16; B (weights) always fp32; C fp32/bf16.
// blockIdx.z batches over (b,h): off = zb*Sb + (zh/aHdiv)*Sh.
// ---------------------------------------------------------------------------
template <bool A_BF16, bool B_NK, bool OUT_BF16>
__global__ __launch_bounds__(256) void gemm_bf16_wmma(
    const void* __restrict__ Ap, const float* __restrict__ Bp,
    const float* __restrict__ bias, void* __restrict__ Cp, int M, int N, int K,
    int lda, int ldb, int ldc, int zdiv, int aHdiv, long long aSb, long long aSh,
    long long bSb, long long bSh, long long cSb, long long cSh) {
  __shared__ bf16_t As[128][32];
  __shared__ bf16_t Bs[128][32];  // always [n][k]

  const int tid = threadIdx.x;
  const int lane = tid & 31;
  const int wid = tid >> 5;
  const int wm = wid >> 1;  // 0..3
  const int wn = wid & 1;   // 0..1

  const int zb = blockIdx.z / zdiv;
  const int zh = blockIdx.z % zdiv;

  const float* Af = nullptr;
  const bf16_t* Ab = nullptr;
  if constexpr (A_BF16)
    Ab = (const bf16_t*)Ap + zb * aSb + (long long)(zh / aHdiv) * aSh;
  else
    Af = (const float*)Ap + zb * aSb + (long long)(zh / aHdiv) * aSh;
  const float* Bw = Bp + zb * bSb + (long long)zh * bSh;
  float* Cf = nullptr;
  bf16_t* Cb = nullptr;
  if constexpr (OUT_BF16)
    Cb = (bf16_t*)Cp + zb * cSb + (long long)zh * cSh;
  else
    Cf = (float*)Cp + zb * cSb + (long long)zh * cSh;

  const int n0 = blockIdx.x * 128;
  const int m0 = blockIdx.y * 128;

  const v8f vzero = {};
  v8f acc[2][4];
#pragma unroll
  for (int i = 0; i < 2; ++i)
#pragma unroll
    for (int j = 0; j < 4; ++j) acc[i][j] = vzero;

  for (int k0 = 0; k0 < K; k0 += 32) {
    // L2 prefetch of next K-slab (lowers to global_prefetch_b8)
    if (k0 + 32 < K) {
      int pr = tid >> 1;  // 128 rows covered by the block
      if constexpr (A_BF16)
        __builtin_prefetch(&Ab[(size_t)(m0 + pr) * lda + k0 + 32], 0, 1);
      else
        __builtin_prefetch(&Af[(size_t)(m0 + pr) * lda + k0 + 32], 0, 1);
      if constexpr (B_NK)
        __builtin_prefetch(&Bw[(size_t)(n0 + pr) * ldb + k0 + 32], 0, 1);
    }
    // Stage A tile (fp32/bf16 -> bf16 LDS)
#pragma unroll
    for (int t = 0; t < 16; ++t) {
      int idx = tid + t * 256;
      int r = idx >> 5, k = idx & 31;
      float v;
      if constexpr (A_BF16)
        v = (float)Ab[(size_t)(m0 + r) * lda + k0 + k];
      else
        v = Af[(size_t)(m0 + r) * lda + k0 + k];
      As[r][k] = (bf16_t)v;
    }
    // Stage B tile as Bs[n][k]
#pragma unroll
    for (int t = 0; t < 16; ++t) {
      int idx = tid + t * 256;
      int n = idx >> 5, k = idx & 31;
      float v;
      if constexpr (B_NK)
        v = Bw[(size_t)(n0 + n) * ldb + k0 + k];
      else
        v = Bw[(size_t)(k0 + k) * ldb + n0 + n];
      Bs[n][k] = (bf16_t)v;
    }
    __syncthreads();

    v16bf afrag[2], bfrag[4];
#pragma unroll
    for (int i = 0; i < 2; ++i)
      afrag[i] = load_frag(&As[0][0], 32, wm * 32 + i * 16, 0, lane);
#pragma unroll
    for (int j = 0; j < 4; ++j)
      bfrag[j] = load_frag(&Bs[0][0], 32, wn * 64 + j * 16, 0, lane);
#pragma unroll
    for (int i = 0; i < 2; ++i)
#pragma unroll
      for (int j = 0; j < 4; ++j)
        acc[i][j] = wmma_bf16(afrag[i], bfrag[j], acc[i][j]);
    __syncthreads();
  }

  // Epilogue (C layout: lane holds col = lane&15; rows (lane>>4)*8 + r)
  const int hf = lane >> 4;
  const int cl = lane & 15;
#pragma unroll
  for (int i = 0; i < 2; ++i) {
#pragma unroll
    for (int j = 0; j < 4; ++j) {
      int col = n0 + wn * 64 + j * 16 + cl;
      float bv = bias ? bias[col] : 0.0f;
#pragma unroll
      for (int r = 0; r < 8; ++r) {
        int row = m0 + wm * 32 + i * 16 + hf * 8 + r;
        float v = acc[i][j][r] + bv;
        if constexpr (OUT_BF16)
          Cb[(size_t)row * ldc + col] = (bf16_t)v;
        else
          Cf[(size_t)row * ldc + col] = v;
      }
    }
  }
}

// ---------------------------------------------------------------------------
// Flash attention per (b,h): q_lat[T,128] x k_lat^T -> online softmax ->
// P x v_heads[T,128]. 128 q-rows per block, 16 per wave; 32-key chunks of
// K/V staged in LDS; causal mask applied analytically. K tile staged with
// async LDS DMA when available (pure bf16 copy); V tile staged transposed.
// ---------------------------------------------------------------------------
__global__ __launch_bounds__(256) void mla_flash_attn(
    const bf16_t* __restrict__ qlat,    // [B,H,T,HD]
    const bf16_t* __restrict__ klat,    // [B*T, LAT] (strided per kv head)
    const bf16_t* __restrict__ vheads,  // [B,H,T,OD]
    bf16_t* __restrict__ outb) {        // [B,T,HID]
  __shared__ bf16_t Ks[32][kHD];        // keys  [kk][c]
  __shared__ bf16_t Vs[kOD][32];        // V transposed -> [n=d][k=kk]
  __shared__ bf16_t Pls[8][16][32];     // per-wave P repack buffer

  const int tid = threadIdx.x, lane = tid & 31, w = tid >> 5;
  const int b = blockIdx.z, h = blockIdx.y;
  const int kv = h / (kH / kKV);
  const int q0blk = blockIdx.x * 128;
  const int qr0 = q0blk + w * 16;
  const float scale = 0.088388347648318447f;  // 1/sqrt(128)

  const bf16_t* qh = qlat + (size_t)(b * kH + h) * kT * kHD;
  const bf16_t* vh = vheads + (size_t)(b * kH + h) * kT * kOD;
  const bf16_t* kb = klat + (size_t)b * kT * kLAT + kv * kHD;

  v16bf aq[4];
#pragma unroll
  for (int c = 0; c < 4; ++c) aq[c] = load_frag(qh, kHD, qr0, c * 32, lane);

  const v8f vzero = {};
  v8f oacc[8];
#pragma unroll
  for (int d = 0; d < 8; ++d) oacc[d] = vzero;
  float mrow[8], lrow[8];
#pragma unroll
  for (int i = 0; i < 8; ++i) {
    mrow[i] = -3.0e38f;
    lrow[i] = 0.0f;
  }

  const int hf = lane >> 4, cl = lane & 15;
  const int kend = q0blk + 128;  // causal: keys beyond block's last row unused

  for (int kc = 0; kc < kend; kc += 32) {
#ifdef USE_ASYNC_LDS
    // K tile: 32 rows x 256B = 512 x 16B chunks; async DMA global->LDS.
#pragma unroll
    for (int t = 0; t < 2; ++t) {
      int c = tid + t * 256;                 // chunk id
      int row = c >> 4, col = (c & 15) * 8;  // 8 bf16 = 16B per chunk
      __builtin_amdgcn_global_load_async_to_lds_b128(
          (AS_GLOBAL v4i_t*)(kb + (size_t)(kc + row) * kLAT + col),
          (AS_LOCAL v4i_t*)&Ks[row][col], 0, 0);
    }
#else
#pragma unroll
    for (int t = 0; t < 16; ++t) {  // stage K chunk (scalar fallback)
      int idx = tid + t * 256;
      int kk = idx >> 7, c = idx & 127;
      Ks[kk][c] = kb[(size_t)(kc + kk) * kLAT + c];
    }
#endif
#pragma unroll
    for (int t = 0; t < 16; ++t) {  // stage V chunk, transposed in-flight
      int idx = tid + t * 256;
      int kk = idx >> 7, d = idx & 127;
      Vs[d][kk] = vh[(size_t)(kc + kk) * kOD + d];
    }
#ifdef USE_ASYNC_LDS
    __builtin_amdgcn_s_wait_asynccnt(0);  // own wave's async DMA done
#endif
    __syncthreads();

    // S = q * k^T  (16 q-rows x 32 keys)
    v8f sacc[2];
    sacc[0] = vzero;
    sacc[1] = vzero;
#pragma unroll
    for (int c = 0; c < 4; ++c) {
#pragma unroll
      for (int j = 0; j < 2; ++j) {
        v16bf bk = load_frag(&Ks[0][0], kHD, j * 16, c * 32, lane);
        sacc[j] = wmma_bf16(aq[c], bk, sacc[j]);
      }
    }

    // Online softmax: each 16-lane half holds one row group -> width-16 shfl.
#pragma unroll
    for (int i = 0; i < 8; ++i) {
      int qrow = qr0 + hf * 8 + i;
      float s0 = sacc[0][i] * scale;
      float s1 = sacc[1][i] * scale;
      if (kc + cl > qrow) s0 = -3.0e38f;
      if (kc + 16 + cl > qrow) s1 = -3.0e38f;
      float mx = fmaxf(s0, s1);
#pragma unroll
      for (int off = 1; off < 16; off <<= 1)
        mx = fmaxf(mx, __shfl_xor(mx, off, 16));
      float mnew = fmaxf(mrow[i], mx);
      float corr = __expf(mrow[i] - mnew);
      float p0 = __expf(s0 - mnew);
      float p1 = __expf(s1 - mnew);
      float ps = p0 + p1;
#pragma unroll
      for (int off = 1; off < 16; off <<= 1) ps += __shfl_xor(ps, off, 16);
      lrow[i] = lrow[i] * corr + ps;
      mrow[i] = mnew;
#pragma unroll
      for (int d = 0; d < 8; ++d) oacc[d][i] *= corr;
      // Repack P (C-layout) -> row-major LDS for A-fragment reload
      Pls[w][hf * 8 + i][cl] = (bf16_t)p0;
      Pls[w][hf * 8 + i][16 + cl] = (bf16_t)p1;
    }

    // O += P * V (wave-private LDS round trip; DS ops in-order per wave)
    v16bf pa = load_frag(&Pls[w][0][0], 32, 0, 0, lane);
#pragma unroll
    for (int d = 0; d < 8; ++d) {
      v16bf bv = load_frag(&Vs[0][0], 32, d * 16, 0, lane);
      oacc[d] = wmma_bf16(pa, bv, oacc[d]);
    }
    __syncthreads();
  }

  // Normalize and scatter into [B,T,H*OD] for the output projection
#pragma unroll
  for (int d = 0; d < 8; ++d) {
#pragma unroll
    for (int i = 0; i < 8; ++i) {
      int qrow = qr0 + hf * 8 + i;
      float v = oacc[d][i] / lrow[i];
      outb[(size_t)(b * kT + qrow) * kHID + h * kOD + d * 16 + cl] = (bf16_t)v;
    }
  }
}

// ---------------------------------------------------------------------------
extern "C" void kernel_launch(void* const* d_in, const int* in_sizes, int n_in,
                              void* d_out, int out_size, void* d_ws,
                              size_t ws_size, hipStream_t stream) {
  (void)in_sizes; (void)n_in; (void)out_size; (void)ws_size;
  const float* x     = (const float*)d_in[0];
  // d_in[1] = attention_mask: causal tril, applied analytically in the kernel
  const float* W_q   = (const float*)d_in[2];
  const float* b_q   = (const float*)d_in[3];
  const float* W_k   = (const float*)d_in[4];
  const float* b_k   = (const float*)d_in[5];
  const float* W_v   = (const float*)d_in[6];
  const float* b_v   = (const float*)d_in[7];
  const float* W_kup = (const float*)d_in[8];
  const float* W_vup = (const float*)d_in[9];
  const float* W_o   = (const float*)d_in[10];
  const float* b_o   = (const float*)d_in[11];
  float* out = (float*)d_out;

  const int M = kB * kT;  // 4096 tokens
  char* ws = (char*)d_ws;
  bf16_t* q_full = (bf16_t*)ws; ws += (size_t)M * kHID * 2;
  bf16_t* k_lat  = (bf16_t*)ws; ws += (size_t)M * kLAT * 2;
  bf16_t* v_lat  = (bf16_t*)ws; ws += (size_t)M * kLAT * 2;
  bf16_t* q_lat  = (bf16_t*)ws; ws += (size_t)kB * kH * kT * kHD * 2;
  bf16_t* v_hd   = (bf16_t*)ws; ws += (size_t)kB * kH * kT * kOD * 2;
  bf16_t* a_out  = (bf16_t*)ws; ws += (size_t)M * kHID * 2;

  dim3 blk(256);

  // 1) q_full = x @ W_q^T + b_q            [4096,2048]
  gemm_bf16_wmma<false, true, true><<<dim3(kHID / 128, M / 128, 1), blk, 0,
                                      stream>>>(
      x, W_q, b_q, q_full, M, kHID, kHID, kHID, kHID, kHID, 1, 1, 0, 0, 0, 0,
      0, 0);
  // 2) k_lat = x @ W_k^T + b_k             [4096,512]
  gemm_bf16_wmma<false, true, true><<<dim3(kLAT / 128, M / 128, 1), blk, 0,
                                      stream>>>(
      x, W_k, b_k, k_lat, M, kLAT, kHID, kHID, kHID, kLAT, 1, 1, 0, 0, 0, 0, 0,
      0);
  // 3) v_lat = x @ W_v^T + b_v             [4096,512]
  gemm_bf16_wmma<false, true, true><<<dim3(kLAT / 128, M / 128, 1), blk, 0,
                                      stream>>>(
      x, W_v, b_v, v_lat, M, kLAT, kHID, kHID, kHID, kLAT, 1, 1, 0, 0, 0, 0, 0,
      0);
  // 4) q_lat[b,h] = q_full[b,:,h*OD:] @ k_up[h]   (B [K,N] form)
  gemm_bf16_wmma<true, false, true><<<dim3(kHD / 128, kT / 128, kB * kH), blk,
                                      0, stream>>>(
      q_full, W_kup, nullptr, q_lat, kT, kHD, kOD, kHID, kHD, kHD, kH, 1,
      (long long)kT * kHID, (long long)kOD, 0, (long long)kOD * kHD,
      (long long)kH * kT * kHD, (long long)kT * kHD);
  // 5) v_hd[b,h] = v_lat[b,:,kv*HD:] @ v_up[h]^T  (B [N,K] form, kv = h/g)
  gemm_bf16_wmma<true, true, true><<<dim3(kOD / 128, kT / 128, kB * kH), blk,
                                     0, stream>>>(
      v_lat, W_vup, nullptr, v_hd, kT, kOD, kHD, kLAT, kHD, kOD, kH, kH / kKV,
      (long long)kT * kLAT, (long long)kHD, 0, (long long)kOD * kHD,
      (long long)kH * kT * kOD, (long long)kT * kOD);
  // 6) causal flash attention -> a_out [B,T,HID]
  mla_flash_attn<<<dim3(kT / 128, kH, kB), blk, 0, stream>>>(q_lat, k_lat,
                                                             v_hd, a_out);
  // 7) out = a_out @ W_o^T + b_o (fp32 output)
  gemm_bf16_wmma<true, true, false><<<dim3(kHID / 128, M / 128, 1), blk, 0,
                                      stream>>>(
      a_out, W_o, b_o, out, M, kHID, kHID, kHID, kHID, kHID, 1, 1, 0, 0, 0, 0,
      0, 0);
}
